// CLIPAttention_10256381903596
// MI455X (gfx1250) — compile-verified, matrix operand order checked by reading, not run on hardware
//
#include <hip/hip_runtime.h>
#include <hip/hip_bf16.h>

typedef __attribute__((ext_vector_type(16))) _Float16 v16h;
typedef __attribute__((ext_vector_type(8)))  float    v8f;

// Problem constants
constexpr int kB = 16;
constexpr int kT = 577;
constexpr int kE = 1024;
constexpr int kH = 16;
constexpr int kD = 64;
constexpr int kS = 654;            // 77 + 577
constexpr int kM = kB * kT;        // 9232 rows, divisible by 16
constexpr int kSP = 672;           // padded S (21 * 32)
constexpr long long OUT_ELEMS = (long long)kB * kT * kE;        // 9,453,568
constexpr long long KV_ELEMS  = (long long)kB * kH * kS * kD;   // 10,715,136
constexpr float NEG_BIG = -1e30f;

// ---------------------------------------------------------------------------
// CDNA5 async copy: global -> LDS, tracked by ASYNCcnt
// ---------------------------------------------------------------------------
__device__ __forceinline__ void async_b128(unsigned int ldsAddr, unsigned long long gAddr) {
    asm volatile("global_load_async_to_lds_b128 %0, %1, off"
                 :: "v"(ldsAddr), "v"(gAddr) : "memory");
}
__device__ __forceinline__ void wait_async0() {
    asm volatile("s_wait_asynccnt 0x0" ::: "memory");
}

// ---------------------------------------------------------------------------
// fp32 -> f16 conversion (vectorized by 4)
// ---------------------------------------------------------------------------
__global__ void f2h4_kernel(const float* __restrict__ s, _Float16* __restrict__ d, int n4) {
    int i = blockIdx.x * blockDim.x + threadIdx.x;
    if (i < n4) {
        float4 f = ((const float4*)s)[i];
        union { _Float16 h[4]; unsigned long long u; } p;
        p.h[0] = (_Float16)f.x; p.h[1] = (_Float16)f.y;
        p.h[2] = (_Float16)f.z; p.h[3] = (_Float16)f.w;
        ((unsigned long long*)d)[i] = p.u;
    }
}

// ---------------------------------------------------------------------------
// textual_kv * w -> K/V output rows [0..76];  textual_kv layout [2, B, H, 77, D]
// ---------------------------------------------------------------------------
__global__ void textual_kernel(const float* __restrict__ tkv, const float* __restrict__ wptr,
                               float* __restrict__ kout, float* __restrict__ vout) {
    int i = blockIdx.x * blockDim.x + threadIdx.x;
    const int N = 2 * kB * kH * 77 * kD;
    if (i >= N) return;
    float w = wptr[0];
    int d = i & 63;
    int r = i >> 6;
    int s = r % 77; r /= 77;
    int h = r & 15; r >>= 4;
    int b = r & 15;
    int part = r >> 4;
    float* dst = part ? vout : kout;
    dst[(((size_t)(b * kH + h)) * kS + s) * kD + d] = w * tkv[i];
}

// ---------------------------------------------------------------------------
// Tiled WMMA GEMM: C[m,n] = sum_k A[m,k] * W[n,k] (+ bias), M=9232, N=K=1024
// 256 threads = 8 waves as 4(m) x 2(n); output tile 64x64, K-stage 64,
// double-buffered LDS filled by global_load_async_to_lds_b128 (ASYNCcnt).
// MODE 0: q -> f16 ws [B,H,T,D] with *0.125 scale
// MODE 1/2: k/v -> fp32 d_out at [bh, 77+t, d]
// MODE 3: o -> fp32 d_out [B,T,E]
// ---------------------------------------------------------------------------
template<int MODE>
__global__ __launch_bounds__(256) void gemm_kernel(
    const _Float16* __restrict__ A,      // [kM, kE] f16
    const _Float16* __restrict__ W,      // [kE, kE] f16 (row n, contiguous k)
    const float*    __restrict__ bias,   // [kE]
    float*          __restrict__ dst_f,
    _Float16*       __restrict__ dst_h)
{
    // 64 rows x 64 halves per stage, padded row stride 72 halves (144 B)
    __shared__ __align__(16) _Float16 As[2][64 * 72];
    __shared__ __align__(16) _Float16 Bs[2][64 * 72];

    const int tid  = threadIdx.x;
    const int m0   = blockIdx.x * 64;
    const int n0   = blockIdx.y * 64;
    const int lane = tid & 31;
    const int wid  = tid >> 5;
    const int mi   = wid & 3;        // 0..3 (m sub-tile)
    const int nj   = wid >> 2;       // 0..1 (n sub-tile pair)
    const int tl   = lane & 15;
    const int hf   = lane >> 4;      // lane half

    // per-thread async-loader precompute: 2 A-segments + 2 B-segments per stage
    unsigned long long aAddr[2], bAddr[2];
    unsigned int       segLds[2];
    #pragma unroll
    for (int j = 0; j < 2; j++) {
        const int seg = tid + 256 * j;      // 0..511
        const int row = seg >> 3;           // 0..63
        const int off = (seg & 7) * 8;      // 0..56 halves
        int gr = m0 + row; if (gr >= kM) gr = kM - 1;
        aAddr[j] = (unsigned long long)(uintptr_t)(A + (size_t)gr * kE + off);
        bAddr[j] = (unsigned long long)(uintptr_t)(W + (size_t)(n0 + row) * kE + off);
        segLds[j] = (unsigned int)(row * 72 + off) * 2u;   // byte offset in tile
    }
    const unsigned int asB[2] = { (unsigned int)(uintptr_t)&As[0][0],
                                  (unsigned int)(uintptr_t)&As[1][0] };
    const unsigned int bsB[2] = { (unsigned int)(uintptr_t)&Bs[0][0],
                                  (unsigned int)(uintptr_t)&Bs[1][0] };

    auto issue = [&](int kb, int buf) {
        #pragma unroll
        for (int j = 0; j < 2; j++) {
            async_b128(asB[buf] + segLds[j], aAddr[j] + (unsigned long long)kb * 2u);
            async_b128(bsB[buf] + segLds[j], bAddr[j] + (unsigned long long)kb * 2u);
        }
    };

    v8f c0 = {}, c1 = {};

    issue(0, 0);
    wait_async0();
    __syncthreads();

    for (int kb = 0; kb < kE; kb += 64) {
        const int buf = (kb >> 6) & 1;
        if (kb + 64 < kE) issue(kb + 64, buf ^ 1);     // prefetch next stage

        const _Float16* Asb = As[buf];
        const _Float16* Bsb = Bs[buf];
        #pragma unroll
        for (int kk = 0; kk < 64; kk += 32) {
            union { uint4 q[2]; v16h v; } a, b0, b1;
            const _Float16* ar  = Asb + (mi * 16 + tl) * 72 + kk;
            a.q[0]  = *(const uint4*)(ar + 8 * hf);
            a.q[1]  = *(const uint4*)(ar + 16 + 8 * hf);
            const _Float16* br0 = Bsb + (nj * 32 + tl) * 72 + kk;
            const _Float16* br1 = Bsb + (nj * 32 + 16 + tl) * 72 + kk;
            b0.q[0] = *(const uint4*)(br0 + 8 * hf);
            b0.q[1] = *(const uint4*)(br0 + 16 + 8 * hf);
            b1.q[0] = *(const uint4*)(br1 + 8 * hf);
            b1.q[1] = *(const uint4*)(br1 + 16 + 8 * hf);
            c0 = __builtin_amdgcn_wmma_f32_16x16x32_f16(false, a.v, false, b0.v, (short)0, c0, false, false);
            c1 = __builtin_amdgcn_wmma_f32_16x16x32_f16(false, a.v, false, b1.v, (short)0, c1, false, false);
        }

        wait_async0();       // my next-stage DMA writes have landed in LDS
        __syncthreads();     // everyone's writes landed, everyone done reading
    }

    #pragma unroll
    for (int sub = 0; sub < 2; sub++) {
        v8f c = sub ? c1 : c0;
        const int n = n0 + nj * 32 + sub * 16 + tl;
        const float bval = bias[n];
        #pragma unroll
        for (int r = 0; r < 8; r++) {
            const int m = m0 + mi * 16 + r + 8 * hf;
            if (m >= kM) continue;
            float val = c[r] + bval;
            if (MODE == 0) {
                val *= 0.125f;  // D^-0.5
                const int b_ = m / kT, t = m % kT;
                const int h_ = n >> 6, d_ = n & 63;
                dst_h[(((size_t)(b_ * kH + h_)) * kT + t) * kD + d_] = (_Float16)val;
            } else if (MODE == 1 || MODE == 2) {
                const int b_ = m / kT, t = m % kT;
                const int h_ = n >> 6, d_ = n & 63;
                dst_f[(((size_t)(b_ * kH + h_)) * kS + 77 + t) * kD + d_] = val;
            } else {
                dst_f[(size_t)m * kE + n] = val;
            }
        }
    }
}

// ---------------------------------------------------------------------------
// Attention: one workgroup per (bh, 16-query tile). 128 threads = 4 waves.
// Pass1: scores = Q K^T into LDS (WMMA). Pass2: softmax in LDS (probs f16).
// Pass3: out = P V (WMMA), rows rescaled by 1/rowsum, -> attn f16 ws [B,T,E].
// ---------------------------------------------------------------------------
__global__ __launch_bounds__(128) void attn_kernel(
    const _Float16* __restrict__ Q,   // [B*H, T, D] f16 (pre-scaled)
    const float*    __restrict__ Kf,  // [B*H, S, D] fp32
    const float*    __restrict__ Vf,  // [B*H, S, D] fp32
    _Float16*       __restrict__ O)   // [B, T, E] f16
{
    __shared__ __align__(16) float    sc[16 * kSP];
    __shared__ __align__(16) _Float16 pr[16 * kSP];
    __shared__ float red[16 * 8];
    __shared__ float rinv[16];

    const int t0   = blockIdx.x;            // 0..36
    const int bh   = blockIdx.y;            // 0..255
    const int b_   = bh >> 4, h_ = bh & 15;
    const int tid  = threadIdx.x;
    const int lane = tid & 31;
    const int wid  = tid >> 5;              // 0..3
    const int tl   = lane & 15;
    const int hf   = lane >> 4;

    // init pad columns [656..671] to -big
    for (int i = tid; i < 16 * 16; i += 128)
        sc[(i >> 4) * kSP + 656 + (i & 15)] = NEG_BIG;

    // Q fragment: fixed per wave-lane (row t, K split over d)
    int tq = t0 * 16 + tl; if (tq > kT - 1) tq = kT - 1;
    const _Float16* qr = Q + ((size_t)bh * kT + tq) * kD;
    union V16 { uint4 q[2]; v16h v; };
    V16 a0, a1;
    a0.q[0] = *(const uint4*)(qr + 8 * hf);
    a0.q[1] = *(const uint4*)(qr + 16 + 8 * hf);
    a1.q[0] = *(const uint4*)(qr + 32 + 8 * hf);
    a1.q[1] = *(const uint4*)(qr + 48 + 8 * hf);

    // Pass 1: scores
    for (int st = wid; st < 41; st += 4) {
        const int s = st * 16 + tl;
        const bool valid = (s < kS);
        const int scl = valid ? s : (kS - 1);
        const float* kr = Kf + ((size_t)bh * kS + scl) * kD;
        union { _Float16 h[16]; v16h v; } b0, b1;
        #pragma unroll
        for (int i = 0; i < 8; i++) {
            b0.h[i]     = (_Float16)kr[      8 * hf + i];
            b0.h[i + 8] = (_Float16)kr[16  + 8 * hf + i];
            b1.h[i]     = (_Float16)kr[32  + 8 * hf + i];
            b1.h[i + 8] = (_Float16)kr[48  + 8 * hf + i];
        }
        v8f c = {};
        c = __builtin_amdgcn_wmma_f32_16x16x32_f16(false, a0.v, false, b0.v, (short)0, c, false, false);
        c = __builtin_amdgcn_wmma_f32_16x16x32_f16(false, a1.v, false, b1.v, (short)0, c, false, false);
        #pragma unroll
        for (int r = 0; r < 8; r++)
            sc[(r + 8 * hf) * kSP + st * 16 + tl] = valid ? c[r] : NEG_BIG;
    }
    __syncthreads();

    // Pass 2: softmax (16 rows x 8 threads each)
    const int row = tid >> 3, sub = tid & 7;
    float lmax = NEG_BIG;
    for (int cix = sub; cix < kSP; cix += 8)
        lmax = fmaxf(lmax, sc[row * kSP + cix]);
    red[row * 8 + sub] = lmax;
    __syncthreads();
    float rmax = NEG_BIG;
    #pragma unroll
    for (int i = 0; i < 8; i++) rmax = fmaxf(rmax, red[row * 8 + i]);
    __syncthreads();
    float lsum = 0.f;
    for (int cix = sub; cix < kSP; cix += 8) {
        float p = __expf(sc[row * kSP + cix] - rmax);
        lsum += p;
        pr[row * kSP + cix] = (_Float16)p;
    }
    red[row * 8 + sub] = lsum;
    __syncthreads();
    if (sub == 0) {
        float tot = 0.f;
        #pragma unroll
        for (int i = 0; i < 8; i++) tot += red[row * 8 + i];
        rinv[row] = 1.0f / tot;
    }
    __syncthreads();

    // Pass 3: P @ V  (each wave handles 16 of the 64 d-columns)
    const int dcol = wid * 16 + tl;
    const float* vb = Vf + (size_t)bh * kS * kD + dcol;
    v8f acc = {};
    for (int kk = 0; kk < kSP; kk += 32) {
        V16 a;
        const _Float16* prow = pr + tl * kSP + kk;
        a.q[0] = *(const uint4*)(prow + 8 * hf);
        a.q[1] = *(const uint4*)(prow + 16 + 8 * hf);
        union { _Float16 h[16]; v16h v; } bf;
        #pragma unroll
        for (int i = 0; i < 8; i++) {
            int s0 = kk + 8 * hf + i;        if (s0 > kS - 1) s0 = kS - 1;
            int s1 = kk + 16 + 8 * hf + i;   if (s1 > kS - 1) s1 = kS - 1;
            bf.h[i]     = (_Float16)vb[(size_t)s0 * kD];
            bf.h[i + 8] = (_Float16)vb[(size_t)s1 * kD];
        }
        acc = __builtin_amdgcn_wmma_f32_16x16x32_f16(false, a.v, false, bf.v, (short)0, acc, false, false);
    }
    #pragma unroll
    for (int r = 0; r < 8; r++) {
        const int mrow = r + 8 * hf;
        const int t = t0 * 16 + mrow;
        if (t < kT)
            O[((size_t)b_ * kT + t) * kE + h_ * kD + dcol] = (_Float16)(acc[r] * rinv[mrow]);
    }
}

// ---------------------------------------------------------------------------
// Launcher
// ---------------------------------------------------------------------------
extern "C" void kernel_launch(void* const* d_in, const int* in_sizes, int n_in,
                              void* d_out, int out_size, void* d_ws, size_t ws_size,
                              hipStream_t stream) {
    const float* hs  = (const float*)d_in[0];
    const float* tkv = (const float*)d_in[1];
    const float* Wq  = (const float*)d_in[2];
    const float* bq  = (const float*)d_in[3];
    const float* Wk  = (const float*)d_in[4];
    const float* bk  = (const float*)d_in[5];
    const float* Wv  = (const float*)d_in[6];
    const float* bv  = (const float*)d_in[7];
    const float* Wo  = (const float*)d_in[8];
    const float* bo  = (const float*)d_in[9];
    const float* tw  = (const float*)d_in[10];

    float* out  = (float*)d_out;
    float* kout = out + OUT_ELEMS;
    float* vout = kout + KV_ELEMS;

    // workspace layout (all 256-aligned)
    char* ws = (char*)d_ws;
    _Float16* hs_h   = (_Float16*)(ws);                       // 18,907,136 B
    _Float16* wq_h   = (_Float16*)(ws + 18907136);            // 4 x 2,097,152 B
    _Float16* wk_h   = wq_h + (size_t)kE * kE;
    _Float16* wv_h   = wk_h + (size_t)kE * kE;
    _Float16* wo_h   = wv_h + (size_t)kE * kE;
    _Float16* q_h    = (_Float16*)(ws + 27295744);            // 18,907,136 B
    _Float16* attn_h = (_Float16*)(ws + 46202880);            // 18,907,136 B

    // 1) conversions to f16
    {
        int n4 = kM * kE / 4;
        f2h4_kernel<<<(n4 + 255) / 256, 256, 0, stream>>>(hs, hs_h, n4);
        int w4 = kE * kE / 4;
        f2h4_kernel<<<(w4 + 255) / 256, 256, 0, stream>>>(Wq, wq_h, w4);
        f2h4_kernel<<<(w4 + 255) / 256, 256, 0, stream>>>(Wk, wk_h, w4);
        f2h4_kernel<<<(w4 + 255) / 256, 256, 0, stream>>>(Wv, wv_h, w4);
        f2h4_kernel<<<(w4 + 255) / 256, 256, 0, stream>>>(Wo, wo_h, w4);
    }

    // 2) textual K/V (rows 0..76 of output K/V)
    {
        int n = 2 * kB * kH * 77 * kD;
        textual_kernel<<<(n + 255) / 256, 256, 0, stream>>>(tkv, tw, kout, vout);
    }

    // 3) Q/K/V projections
    dim3 ggrid((kM + 63) / 64, kE / 64);   // 145 x 16
    gemm_kernel<0><<<ggrid, 256, 0, stream>>>(hs_h, wq_h, bq, nullptr, q_h);
    gemm_kernel<1><<<ggrid, 256, 0, stream>>>(hs_h, wk_h, bk, kout, nullptr);
    gemm_kernel<2><<<ggrid, 256, 0, stream>>>(hs_h, wv_h, bv, vout, nullptr);

    // 4) attention
    attn_kernel<<<dim3((kT + 15) / 16, kB * kH), 128, 0, stream>>>(q_h, kout, vout, attn_h);

    // 5) output projection
    gemm_kernel<3><<<ggrid, 256, 0, stream>>>(attn_h, wo_h, bo, out, nullptr);
}
